// ConditionalLinearAttention_40793599377749
// MI455X (gfx1250) — compile-verified, matrix operand order checked by reading, not run on hardware
//
#include <hip/hip_runtime.h>
#include <hip/hip_bf16.h>

// ---------------------------------------------------------------------------
// ConditionalLinearAttention for MI455X (gfx1250, wave32, WMMA).
// All GEMM-shaped math runs on v_wmma_f32_16x16x32_bf16 (f32 accumulate).
// Softmax / FiLM / LayerNorm epilogues in f32 VALU.
// ---------------------------------------------------------------------------

typedef __attribute__((ext_vector_type(16))) __bf16        v16bf;
typedef __attribute__((ext_vector_type(8)))  float         v8f;
typedef __attribute__((ext_vector_type(8)))  unsigned int  v8u;

#define WMMA_BF16(A, B, C) \
  __builtin_amdgcn_wmma_f32_16x16x32_bf16(false, (A), false, (B), (short)0, (C), false, false)

__device__ __forceinline__ unsigned short f2bf(float f) {
  unsigned int u = __builtin_bit_cast(unsigned int, f);
  u += 0x7fffu + ((u >> 16) & 1u);            // round-to-nearest-even
  return (unsigned short)(u >> 16);
}

// Load one 16x32 bf16 A/B fragment for wave32 WMMA.
// rowptr points at this lane's matrix row (bf16, K contiguous).
// K pair held in dword v of the frag: k = k0 + (v&4)*4 + (v&3)*2 + 8*half.
__device__ __forceinline__ v16bf frag_ld(const unsigned short* rowptr, int k0, int half) {
  v8u r;
#pragma unroll
  for (int v = 0; v < 8; ++v) {
    int k = k0 + ((v & 4) << 2) + ((v & 3) << 1) + (half << 3);
    r[v] = *reinterpret_cast<const unsigned int*>(rowptr + k);
  }
  return __builtin_bit_cast(v16bf, r);
}

// ----- problem constants ----------------------------------------------------
#define B_    16
#define DIM_  256
#define NSP_  4096           // h*w
#define HID_  128            // heads*dim_head
#define HEADS_ 4
#define DH_   32

// ----- workspace layout (bytes) --------------------------------------------
static constexpr size_t OFF_WQ   = 0;                         // 384x256 bf16
static constexpr size_t OFF_WO   = OFF_WQ  + 384*256*2;       // 256x128 bf16
static constexpr size_t OFF_COND = OFF_WO  + 256*128*2;       // 16x512  f32
static constexpr size_t OFF_XT   = OFF_COND + 16*512*4;       // b,n,c   bf16
static constexpr size_t OFF_Q    = OFF_XT  + (size_t)B_*NSP_*DIM_*2;   // f32 b,128,n
static constexpr size_t OFF_K    = OFF_Q   + (size_t)B_*HID_*NSP_*4;   // f32 b,128,n
static constexpr size_t OFF_KSM  = OFF_K   + (size_t)B_*HID_*NSP_*4;   // bf16
static constexpr size_t OFF_V    = OFF_KSM + (size_t)B_*HID_*NSP_*2;   // bf16
static constexpr size_t OFF_CTX  = OFF_V   + (size_t)B_*HID_*NSP_*2;   // f32 b,h,32,32
static constexpr size_t OFF_OT   = OFF_CTX + (size_t)B_*HEADS_*DH_*DH_*4; // bf16 b,n,128

// ===========================================================================
// 1) weights fp32 -> bf16
__global__ void prep_weights(const float* __restrict__ wq, const float* __restrict__ wo,
                             unsigned short* __restrict__ wqb, unsigned short* __restrict__ wob) {
  int i = blockIdx.x * 256 + threadIdx.x;
  if (i < 384 * 256) wqb[i] = f2bf(wq[i]);
  if (i < 256 * 128) wob[i] = f2bf(wo[i]);
}

// 2) cond_out = W_cond @ silu(cond) + b_cond    (16 x 512, f32)
__global__ void cond_mlp(const float* __restrict__ cond, const float* __restrict__ Wc,
                         const float* __restrict__ bc, float* __restrict__ co) {
  __shared__ float ca[256];
  const int b = blockIdx.x, t = threadIdx.x;
  float cv = cond[b * 256 + t];
  ca[t] = cv / (1.f + __expf(-cv));          // silu
  __syncthreads();
  for (int j = t; j < 512; j += 256) {
    const float* wr = Wc + (size_t)j * 256;
    float acc = bc[j];
    for (int i = 0; i < 256; ++i) acc += wr[i] * ca[i];
    co[b * 512 + j] = acc;
  }
}

// 3) x (b,256,4096) f32 -> xT (b,4096,256) bf16
__global__ void transpose_x(const float* __restrict__ x, unsigned short* __restrict__ xT) {
  __shared__ float tile[32][33];
  const int b = blockIdx.z, n0 = blockIdx.x * 32, c0 = blockIdx.y * 32;
  const int tx = threadIdx.x, ty = threadIdx.y;          // 32 x 8
  const float* xb = x + (size_t)b * DIM_ * NSP_;
#pragma unroll
  for (int i = 0; i < 4; ++i)
    tile[ty + i * 8][tx] = xb[(size_t)(c0 + ty + i * 8) * NSP_ + n0 + tx];
  __syncthreads();
  unsigned short* xo = xT + (size_t)b * NSP_ * DIM_;
#pragma unroll
  for (int i = 0; i < 4; ++i)
    xo[(size_t)(n0 + ty + i * 8) * DIM_ + c0 + tx] = f2bf(tile[tx][ty + i * 8]);
}

// 4) qkv GEMM (384x4096x256 per batch) + FiLM epilogue.
//    wave -> 32x32 output tile (2x2 WMMA frags), 4 waves/block along N.
__global__ void gemm_qkv_film(const unsigned short* __restrict__ wq,
                              const unsigned short* __restrict__ xT,
                              const float* __restrict__ co,
                              float* __restrict__ qbuf, float* __restrict__ kbuf,
                              unsigned short* __restrict__ vbf) {
  const int b = blockIdx.x;
  const int o0 = blockIdx.y * 32;
  const int wave = threadIdx.x >> 5, lane = threadIdx.x & 31;
  const int n0 = (blockIdx.z * 4 + wave) * 32;
  const int half = lane >> 4, m = lane & 15;
  const unsigned short* xb = xT + (size_t)b * NSP_ * DIM_;
  v8f c00 = {}, c01 = {}, c10 = {}, c11 = {};
#pragma unroll
  for (int k0 = 0; k0 < DIM_; k0 += 32) {
    v16bf a0 = frag_ld(wq + (size_t)(o0 + m)      * DIM_, k0, half);
    v16bf a1 = frag_ld(wq + (size_t)(o0 + 16 + m) * DIM_, k0, half);
    v16bf b0 = frag_ld(xb + (size_t)(n0 + m)      * DIM_, k0, half);
    v16bf b1 = frag_ld(xb + (size_t)(n0 + 16 + m) * DIM_, k0, half);
    c00 = WMMA_BF16(a0, b0, c00);
    c01 = WMMA_BF16(a0, b1, c01);
    c10 = WMMA_BF16(a1, b0, c10);
    c11 = WMMA_BF16(a1, b1, c11);
  }
  const float* cb = co + b * 512;
  auto writeFrag = [&](const v8f& c, int ot, int nt) {
#pragma unroll
    for (int r = 0; r < 8; ++r) {
      int o = ot + r + half * 8;
      size_t n = (size_t)(nt + m);
      float v = c[r];
      if (o < 128) {                                   // q: raw
        qbuf[((size_t)b * HID_ + o) * NSP_ + n] = v;
      } else if (o < 256) {                            // k: FiLM
        int cc = o - 128;
        kbuf[((size_t)b * HID_ + cc) * NSP_ + n] = (1.f + cb[cc]) * v + cb[128 + cc];
      } else {                                         // v: FiLM then /hw
        int cc = o - 256;
        float vv = ((1.f + cb[256 + cc]) * v + cb[384 + cc]) * (1.f / 4096.f);
        vbf[((size_t)b * HID_ + cc) * NSP_ + n] = f2bf(vv);
      }
    }
  };
  writeFrag(c00, o0, n0);       writeFrag(c01, o0, n0 + 16);
  writeFrag(c10, o0 + 16, n0);  writeFrag(c11, o0 + 16, n0 + 16);
}

// 5) k softmax over spatial axis (4096) per (b, channel); output bf16.
__global__ void k_softmax(const float* __restrict__ kbuf, unsigned short* __restrict__ ksm) {
  __shared__ float red[256];
  const int c = blockIdx.x, b = blockIdx.y, t = threadIdx.x;
  const size_t base = ((size_t)b * HID_ + c) * NSP_;
  float v[16], mx = -1e30f;
#pragma unroll
  for (int i = 0; i < 16; ++i) { v[i] = kbuf[base + t + i * 256]; mx = fmaxf(mx, v[i]); }
  red[t] = mx; __syncthreads();
  for (int s = 128; s > 0; s >>= 1) {
    if (t < s) red[t] = fmaxf(red[t], red[t + s]);
    __syncthreads();
  }
  mx = red[0]; __syncthreads();
  float sum = 0.f;
#pragma unroll
  for (int i = 0; i < 16; ++i) { v[i] = __expf(v[i] - mx); sum += v[i]; }
  red[t] = sum; __syncthreads();
  for (int s = 128; s > 0; s >>= 1) {
    if (t < s) red[t] += red[t + s];
    __syncthreads();
  }
  float inv = 1.f / red[0];
#pragma unroll
  for (int i = 0; i < 16; ++i) ksm[base + t + i * 256] = f2bf(v[i] * inv);
}

// 6) context[c][e] = sum_n ksm[c][n] * v[e][n]   (32x32 per (b,head), K=4096)
//    8 waves split-K, WMMA, LDS tree reduce (fixed order -> deterministic).
__global__ void context_k(const unsigned short* __restrict__ ksm,
                          const unsigned short* __restrict__ vbf,
                          float* __restrict__ ctx) {
  __shared__ float part[8 * 1024];
  const int h = blockIdx.x, b = blockIdx.y;
  const int wave = threadIdx.x >> 5, lane = threadIdx.x & 31;
  const int half = lane >> 4, m = lane & 15;
  const unsigned short* kb = ksm + ((size_t)b * HID_ + h * DH_) * NSP_;
  const unsigned short* vb = vbf + ((size_t)b * HID_ + h * DH_) * NSP_;
  v8f c00 = {}, c01 = {}, c10 = {}, c11 = {};
  for (int kc = 0; kc < 16; ++kc) {
    int k0 = wave * 512 + kc * 32;
    v16bf a0 = frag_ld(kb + (size_t)m        * NSP_, k0, half);
    v16bf a1 = frag_ld(kb + (size_t)(16 + m) * NSP_, k0, half);
    v16bf b0 = frag_ld(vb + (size_t)m        * NSP_, k0, half);
    v16bf b1 = frag_ld(vb + (size_t)(16 + m) * NSP_, k0, half);
    c00 = WMMA_BF16(a0, b0, c00);
    c01 = WMMA_BF16(a0, b1, c01);
    c10 = WMMA_BF16(a1, b0, c10);
    c11 = WMMA_BF16(a1, b1, c11);
  }
  auto stash = [&](const v8f& c, int ct, int et) {
#pragma unroll
    for (int r = 0; r < 8; ++r)
      part[wave * 1024 + (ct + r + half * 8) * 32 + (et + m)] = c[r];
  };
  stash(c00, 0, 0);  stash(c01, 0, 16);  stash(c10, 16, 0);  stash(c11, 16, 16);
  __syncthreads();
  for (int idx = threadIdx.x; idx < 1024; idx += 256) {
    float s = 0.f;
#pragma unroll
    for (int w = 0; w < 8; ++w) s += part[w * 1024 + idx];
    ctx[((size_t)(b * HEADS_ + h)) * 1024 + idx] = s;
  }
}

// 7) per-position q channel-softmax (32) * scale, fused with ctx^T @ q.
//    Writes transposed bf16 buffer oT[b][n][hid] for GEMM2 B-fragments.
__global__ void attn_out(const float* __restrict__ qbuf, const float* __restrict__ ctx,
                         unsigned short* __restrict__ oT) {
  __shared__ float cl[1024];
  const int nt = blockIdx.x, h = blockIdx.y, b = blockIdx.z;
  for (int i = threadIdx.x; i < 1024; i += 256)
    cl[i] = ctx[((size_t)(b * HEADS_ + h)) * 1024 + i];
  __syncthreads();
  const int n = nt * 256 + threadIdx.x;
  const float* qp = qbuf + ((size_t)b * HID_ + h * DH_) * NSP_ + n;
  float qv[32], mx = -1e30f;
#pragma unroll
  for (int c = 0; c < 32; ++c) { qv[c] = qp[(size_t)c * NSP_]; mx = fmaxf(mx, qv[c]); }
  float sum = 0.f;
#pragma unroll
  for (int c = 0; c < 32; ++c) { qv[c] = __expf(qv[c] - mx); sum += qv[c]; }
  const float sc = 0.17677669529663687f / sum;     // dim_head^-0.5 / sum
#pragma unroll
  for (int c = 0; c < 32; ++c) qv[c] *= sc;
  unsigned short* op = oT + ((size_t)b * NSP_ + n) * HID_ + h * DH_;
#pragma unroll
  for (int e = 0; e < 32; ++e) {
    float acc = 0.f;
#pragma unroll
    for (int c = 0; c < 32; ++c) acc += cl[c * 32 + e] * qv[c];
    op[e] = f2bf(acc);
  }
}

// 8) out GEMM (256x4096x128 per batch) + bias + channel LayerNorm + g scale.
//    Block owns a full 256(M) x 32(N) tile so LN over channels stays in LDS.
__global__ void gemm_out_ln(const unsigned short* __restrict__ wo,
                            const unsigned short* __restrict__ oT,
                            const float* __restrict__ bo,
                            const float* __restrict__ g,
                            float* __restrict__ out) {
  __shared__ float yt[256 * 33];
  __shared__ float mu[32], rs[32];
  const int n0 = blockIdx.x * 32, b = blockIdx.y;
  const int wave = threadIdx.x >> 5, lane = threadIdx.x & 31;
  const int half = lane >> 4, m = lane & 15;
  const int o0 = wave * 32;
  const unsigned short* ob = oT + (size_t)b * NSP_ * HID_;
  v8f c00 = {}, c01 = {}, c10 = {}, c11 = {};
#pragma unroll
  for (int k0 = 0; k0 < HID_; k0 += 32) {
    v16bf a0 = frag_ld(wo + (size_t)(o0 + m)      * HID_, k0, half);
    v16bf a1 = frag_ld(wo + (size_t)(o0 + 16 + m) * HID_, k0, half);
    v16bf b0 = frag_ld(ob + (size_t)(n0 + m)      * HID_, k0, half);
    v16bf b1 = frag_ld(ob + (size_t)(n0 + 16 + m) * HID_, k0, half);
    c00 = WMMA_BF16(a0, b0, c00);
    c01 = WMMA_BF16(a0, b1, c01);
    c10 = WMMA_BF16(a1, b0, c10);
    c11 = WMMA_BF16(a1, b1, c11);
  }
  auto stash = [&](const v8f& c, int ot, int nt) {
#pragma unroll
    for (int r = 0; r < 8; ++r) {
      int o = ot + r + half * 8;
      yt[o * 33 + nt + m] = c[r] + bo[o];
    }
  };
  stash(c00, o0, 0);  stash(c01, o0, 16);  stash(c10, o0 + 16, 0);  stash(c11, o0 + 16, 16);
  __syncthreads();
  if (threadIdx.x < 32) {
    int j = threadIdx.x;
    float s = 0.f, s2 = 0.f;
    for (int o = 0; o < 256; ++o) { float v = yt[o * 33 + j]; s += v; s2 += v * v; }
    float mean = s * (1.f / 256.f);
    float var  = s2 * (1.f / 256.f) - mean * mean;   // biased, matches jnp.var
    mu[j] = mean;
    rs[j] = rsqrtf(var + 1e-5f);
  }
  __syncthreads();
  for (int idx = threadIdx.x; idx < 256 * 32; idx += 256) {
    int o = idx >> 5, j = idx & 31;
    out[((size_t)b * DIM_ + o) * NSP_ + n0 + j] = (yt[o * 33 + j] - mu[j]) * rs[j] * g[o];
  }
}

// ===========================================================================
extern "C" void kernel_launch(void* const* d_in, const int* in_sizes, int n_in,
                              void* d_out, int out_size, void* d_ws, size_t ws_size,
                              hipStream_t stream) {
  const float* x      = (const float*)d_in[0];
  const float* cond   = (const float*)d_in[1];
  const float* W_qkv  = (const float*)d_in[2];
  const float* W_cond = (const float*)d_in[3];
  const float* b_cond = (const float*)d_in[4];
  const float* W_out  = (const float*)d_in[5];
  const float* b_out  = (const float*)d_in[6];
  const float* g      = (const float*)d_in[7];
  float* out = (float*)d_out;

  char* ws = (char*)d_ws;
  unsigned short* wqb  = (unsigned short*)(ws + OFF_WQ);
  unsigned short* wob  = (unsigned short*)(ws + OFF_WO);
  float*          co   = (float*)         (ws + OFF_COND);
  unsigned short* xT   = (unsigned short*)(ws + OFF_XT);
  float*          qbuf = (float*)         (ws + OFF_Q);
  float*          kbuf = (float*)         (ws + OFF_K);
  unsigned short* ksm  = (unsigned short*)(ws + OFF_KSM);
  unsigned short* vbf  = (unsigned short*)(ws + OFF_V);
  float*          ctx  = (float*)         (ws + OFF_CTX);
  unsigned short* oT   = (unsigned short*)(ws + OFF_OT);

  prep_weights <<<384, 256, 0, stream>>>(W_qkv, W_out, wqb, wob);
  cond_mlp     <<<B_, 256, 0, stream>>>(cond, W_cond, b_cond, co);
  transpose_x  <<<dim3(NSP_ / 32, DIM_ / 32, B_), dim3(32, 8), 0, stream>>>(x, xT);
  gemm_qkv_film<<<dim3(B_, 12, 32), 128, 0, stream>>>(wqb, xT, co, qbuf, kbuf, vbf);
  k_softmax    <<<dim3(HID_, B_), 256, 0, stream>>>(kbuf, ksm);
  context_k    <<<dim3(HEADS_, B_), 256, 0, stream>>>(ksm, vbf, ctx);
  attn_out     <<<dim3(NSP_ / 256, HEADS_, B_), 256, 0, stream>>>(qbuf, ctx, oT);
  gemm_out_ln  <<<dim3(NSP_ / 32, B_), 256, 0, stream>>>(wob, oT, b_out, g, out);
}
